// Linear_2018634629251
// MI455X (gfx1250) — matrix operand order repaired and checked by reading, not compiled, not run on hardware
//
#include <hip/hip_runtime.h>

// GPTQ uint4 dequant + GEMM for gfx1250 (MI455X), wave32 WMMA f16->f32.
// out[m,n] = sum_k A[m,k] * (q[n,k]-z[n,g])*s[n,g] + bias[n]
// Double-buffered LDS pipeline: global loads for tile i+1 issued before the
// WMMAs of tile i so vector-memory latency hides behind the matrix pipe.

typedef __attribute__((ext_vector_type(16))) _Float16     v16h;
typedef __attribute__((ext_vector_type(4)))  _Float16     v4h;
typedef __attribute__((ext_vector_type(8)))  float        v8f;
typedef __attribute__((ext_vector_type(4)))  unsigned int v4u;

#define BM 128
#define BN 128
#define BK 32
#define LDP 40   // padded LDS row stride in halves: 80B = 5*16B -> aligned b128
                 // loads AND bank-conflict-free row gathers (20*r mod 64 perm)

union Frag { v4u u[2]; v16h v; };

__global__ __launch_bounds__(256)
void gptq_gemm_wmma(const float* __restrict__ A,
                    const int*   __restrict__ Q,
                    const float* __restrict__ S,
                    const float* __restrict__ Z,
                    const float* __restrict__ bias,
                    float*       __restrict__ out,
                    int M, int N, int K, int G, int gsh)
{
    __shared__ _Float16 As[2][BM][LDP];   // A tiles, f16 (double buffered)
    __shared__ _Float16 Ws[2][BN][LDP];   // dequantized W tiles, f16

    const int t    = threadIdx.x;
    const int lane = t & 31;
    const int wave = t >> 5;
    const int wm   = (wave & 1) * 64;    // wave M offset within block tile
    const int wn   = (wave >> 1) * 32;   // wave N offset within block tile
    const int lmod = lane & 15;
    const int lhi  = lane >> 4;

    const int m0 = blockIdx.y * BM;
    const int n0 = blockIdx.x * BN;

    // cooperative-load geometry: thread owns 4 float4/int4 slots, fixed column
    const int rw = t >> 3;               // base row slice 0..31
    const int c4 = t & 7;                // float4 column within BK

    const float* ap[4]; const int* qp[4]; int nrow[4];
    #pragma unroll
    for (int j = 0; j < 4; ++j) {
        const int row = rw + j * 32;
        ap[j]   = A + (size_t)(m0 + row) * K + c4 * 4;
        qp[j]   = Q + (size_t)(n0 + row) * K + c4 * 4;
        nrow[j] = n0 + row;
    }

    v8f acc[4][2] = {};

    // ---- pipeline prologue: tile 0 -> registers ----
    float4 ar[4]; int4 qr[4]; float sr[4], zr[4];
    #pragma unroll
    for (int j = 0; j < 4; ++j) {
        ar[j] = *(const float4*)(ap[j]);
        qr[j] = *(const int4*)  (qp[j]);
        sr[j] = S[(size_t)nrow[j] * G];   // group 0
        zr[j] = Z[(size_t)nrow[j] * G];
    }

    int buf = 0;
    for (int kk = 0; kk < K; kk += BK) {
        // ---- stage registers -> LDS[buf] (f32->f16 convert + dequant) ----
        #pragma unroll
        for (int j = 0; j < 4; ++j) {
            const int row = rw + j * 32;
            v4h ha;
            ha.x = (_Float16)ar[j].x; ha.y = (_Float16)ar[j].y;
            ha.z = (_Float16)ar[j].z; ha.w = (_Float16)ar[j].w;
            *(v4h*)&As[buf][row][c4 * 4] = ha;
            v4h hw;
            hw.x = (_Float16)(((float)qr[j].x - zr[j]) * sr[j]);
            hw.y = (_Float16)(((float)qr[j].y - zr[j]) * sr[j]);
            hw.z = (_Float16)(((float)qr[j].z - zr[j]) * sr[j]);
            hw.w = (_Float16)(((float)qr[j].w - zr[j]) * sr[j]);
            *(v4h*)&Ws[buf][row][c4 * 4] = hw;
        }
        __syncthreads();   // LDS[buf] visible to all waves

        // ---- issue NEXT tile's global loads (in flight during WMMAs) ----
        if (kk + BK < K) {
            const int g = (kk + BK) >> gsh;   // group index, power-of-2 gs
            #pragma unroll
            for (int j = 0; j < 4; ++j) {
                ar[j] = *(const float4*)(ap[j] + kk + BK);
                qr[j] = *(const int4*)  (qp[j] + kk + BK);
                sr[j] = S[(size_t)nrow[j] * G + g];
                zr[j] = Z[(size_t)nrow[j] * G + g];
            }
        }

        // ---- fragments (ISA VGPR layouts) + 8 WMMAs from LDS[buf] ----
        Frag a[4], b[2];
        #pragma unroll
        for (int mt = 0; mt < 4; ++mt) {
            // A 16x32: lane holds row m=lane&15; K chunks (lhi*8..+7) and +16
            const _Float16* p = &As[buf][wm + mt * 16 + lmod][lhi * 8];
            a[mt].u[0] = *(const v4u*)p;         // halves kb+0..7
            a[mt].u[1] = *(const v4u*)(p + 16);  // halves kb+16..23
        }
        #pragma unroll
        for (int nt = 0; nt < 2; ++nt) {
            // B 32x16: lane holds column n=lane&15; K chunks (lhi*8..+7) and +16
            const _Float16* p = &Ws[buf][wn + nt * 16 + lmod][lhi * 8];
            b[nt].u[0] = *(const v4u*)p;         // halves kb+0..7
            b[nt].u[1] = *(const v4u*)(p + 16);  // halves kb+16..23
        }
        #pragma unroll
        for (int mt = 0; mt < 4; ++mt)
            #pragma unroll
            for (int nt = 0; nt < 2; ++nt)
                acc[mt][nt] = __builtin_amdgcn_wmma_f32_16x16x32_f16(
                    /*neg_a=*/false, a[mt].v, /*neg_b=*/false, b[nt].v,
                    /*c_mod=*/(short)0, acc[mt][nt],
                    /*reuse_a=*/false, /*reuse_b=*/false);

        buf ^= 1;
        // Single barrier per iteration is sufficient: the next write to this
        // buffer happens after the NEXT iteration's barrier, which all waves
        // only pass after finishing this iteration's reads.
    }

    // ---- epilogue: D layout -> global, fused bias ----
    #pragma unroll
    for (int mt = 0; mt < 4; ++mt) {
        #pragma unroll
        for (int nt = 0; nt < 2; ++nt) {
            const int   n  = n0 + wn + nt * 16 + lmod;
            const float bv = bias[n];
            const int   mb = m0 + wm + mt * 16 + lhi * 8;
            #pragma unroll
            for (int r = 0; r < 8; ++r)
                out[(size_t)(mb + r) * N + n] = acc[mt][nt][r] + bv;
        }
    }
}

extern "C" void kernel_launch(void* const* d_in, const int* in_sizes, int n_in,
                              void* d_out, int out_size, void* d_ws, size_t ws_size,
                              hipStream_t stream) {
    const float* A  = (const float*)d_in[0];
    const int*   Q  = (const int*)  d_in[1];
    const float* S  = (const float*)d_in[2];
    const float* Z  = (const float*)d_in[3];
    const float* Bi = (const float*)d_in[4];
    float* out = (float*)d_out;

    const int N  = in_sizes[4];
    const int G  = in_sizes[2] / N;        // 32 groups
    const int K  = in_sizes[1] / N;        // 4096
    const int M  = in_sizes[0] / K;        // 2048
    const int gs = K / G;                  // 128 (power of two)
    const int gsh = __builtin_ctz(gs);     // 7

    dim3 grid(N / BN, M / BM);             // 32 x 16 blocks
    gptq_gemm_wmma<<<grid, 256, 0, stream>>>(A, Q, S, Z, Bi, out, M, N, K, G, gsh);
}
